// GraphClassifier_47708496724388
// MI455X (gfx1250) — compile-verified, matrix operand order checked by reading, not run on hardware
//
#include <hip/hip_runtime.h>

#define N_NODES 8192
#define DIN 512
#define DH 512
#define DOUT 256
#define NUM_GRAPHS 64
#define NUM_CLASSES 10

typedef __attribute__((ext_vector_type(16))) __bf16 v16bf;
typedef __attribute__((ext_vector_type(8)))  float  f32x8;

__device__ __forceinline__ unsigned short f2bf(float f) {
    union { float f; unsigned u; } x; x.f = f;
    unsigned u = x.u;
    unsigned r = (u + 0x7FFFu + ((u >> 16) & 1u)) >> 16;   // round-to-nearest-even
    return (unsigned short)r;
}

// ---------------------------------------------------------------------------
// K0: per-row degree of (A + I) -> dinv = rsqrt(sum + 1)
// ---------------------------------------------------------------------------
__global__ void rowsum_dinv_kernel(const float* __restrict__ adj,
                                   float* __restrict__ dinv) {
    __shared__ float red[256];
    const int row = blockIdx.x;
    const float* p = adj + (size_t)row * N_NODES;
    float s = 0.f;
    for (int j = threadIdx.x; j < N_NODES; j += 256) s += p[j];
    red[threadIdx.x] = s;
    __syncthreads();
    for (int off = 128; off > 0; off >>= 1) {
        if (threadIdx.x < off) red[threadIdx.x] += red[threadIdx.x + off];
        __syncthreads();
    }
    if (threadIdx.x == 0) dinv[row] = rsqrtf(red[0] + 1.0f);
}

// ---------------------------------------------------------------------------
// K1: A_bf[i][j] = bf16( (adj[i][j] + (i==j)) * dinv[i] * dinv[j] )
//     4 elements / thread (one float4 load, one ushort4 store)
// ---------------------------------------------------------------------------
__global__ void norm_cvt_kernel(const float* __restrict__ adj,
                                const float* __restrict__ dinv,
                                unsigned short* __restrict__ abf) {
    const size_t gid = (size_t)blockIdx.x * blockDim.x + threadIdx.x;
    const size_t e = gid * 4;                       // 4 contiguous elems, same row
    const int i  = (int)(e >> 13);                  // /8192
    const int j0 = (int)(e & (N_NODES - 1));
    const float di = dinv[i];
    const float4 a = *(const float4*)(adj + e);
    ushort4 o;
    o.x = f2bf((a.x + (j0 + 0 == i ? 1.f : 0.f)) * di * dinv[j0 + 0]);
    o.y = f2bf((a.y + (j0 + 1 == i ? 1.f : 0.f)) * di * dinv[j0 + 1]);
    o.z = f2bf((a.z + (j0 + 2 == i ? 1.f : 0.f)) * di * dinv[j0 + 2]);
    o.w = f2bf((a.w + (j0 + 3 == i ? 1.f : 0.f)) * di * dinv[j0 + 3]);
    *(ushort4*)(abf + e) = o;
}

// ---------------------------------------------------------------------------
// f32 -> bf16 (4 elems/thread), and transpose+convert (out[c*R+r] = in[r*C+c])
// ---------------------------------------------------------------------------
__global__ void cvt_bf16_kernel(const float* __restrict__ in,
                                unsigned short* __restrict__ out, int n4) {
    const int gid = blockIdx.x * blockDim.x + threadIdx.x;
    if (gid >= n4) return;
    const float4 a = *(const float4*)(in + (size_t)gid * 4);
    ushort4 o; o.x = f2bf(a.x); o.y = f2bf(a.y); o.z = f2bf(a.z); o.w = f2bf(a.w);
    *(ushort4*)(out + (size_t)gid * 4) = o;
}

__global__ void transpose_cvt_kernel(const float* __restrict__ in,
                                     unsigned short* __restrict__ out,
                                     int R, int C) {
    const int gid = blockIdx.x * blockDim.x + threadIdx.x;
    if (gid >= R * C) return;
    const int r = gid % R;      // inner (contiguous) out dim
    const int c = gid / R;
    out[gid] = f2bf(in[(size_t)r * C + c]);
}

// ---------------------------------------------------------------------------
// bf16 WMMA GEMM:  C = A(MxK, row-major) * Bt(NxK, row-major)^T
//   block tile 128x128, K-step 32, 8 waves (2x4), wave tile 64x32
//   -> 8 x v_wmma_f32_16x16x32_bf16 per wave per K-step
// Epilogue: optional bias (per-N, f32), ReLU, bf16 or f32 out,
//           normal [M][N] or transposed [N][M] store.
// ---------------------------------------------------------------------------
template<bool OUT_BF16, bool TRANSPOSED, bool RELU>
__global__ __launch_bounds__(256)
void gemm_wmma_kernel(const unsigned short* __restrict__ A, int lda,
                      const unsigned short* __restrict__ Bt, int ldb,
                      void* __restrict__ Cp, int ldc,
                      const float* __restrict__ bias, int K) {
    // LDS tiles: [128 rows][32 k] bf16, row stride padded to 40 halves so the
    // per-lane b128 fragment reads perfectly tile all 64 LDS banks.
    __shared__ __align__(16) unsigned short Als[128 * 40];
    __shared__ __align__(16) unsigned short Bls[128 * 40];

    const int tid  = threadIdx.x;
    const int lane = tid & 31;
    const int wave = tid >> 5;
    const int wm   = wave & 1;            // wave row   (0..1) -> 64 rows
    const int wn   = wave >> 1;           // wave col   (0..3) -> 32 cols
    const int bm   = blockIdx.x * 128;
    const int bn   = blockIdx.y * 128;

    f32x8 acc[4][2] = {};

    const int lane16 = lane & 15;
    const int halfk  = (lane >> 4) << 3;  // 0 for lanes 0-15, 8 for lanes 16-31

    for (int k0 = 0; k0 < K; k0 += 32) {
        // ---- stage A and Bt tiles: 16B per thread per chunk, 2 chunks each
        #pragma unroll
        for (int it = 0; it < 2; ++it) {
            const int c   = tid + it * 256;
            const int row = c >> 2;
            const int seg = c & 3;
            const unsigned short* ga = A  + (size_t)(bm + row) * lda + k0 + seg * 8;
            const unsigned short* gb = Bt + (size_t)(bn + row) * ldb + k0 + seg * 8;
            *(uint4*)&Als[row * 40 + seg * 8] = *(const uint4*)ga;
            *(uint4*)&Bls[row * 40 + seg * 8] = *(const uint4*)gb;
            if (k0 + 32 < K) {                       // prefetch next K tile
                __builtin_prefetch(ga + 32, 0, 1);
                __builtin_prefetch(gb + 32, 0, 1);
            }
        }
        __syncthreads();

        // ---- fragments: two 16B LDS reads each (ISA 16-bit A layout:
        //      lanes 0-15 hold K 0-7 & 16-23, lanes 16-31 hold K 8-15 & 24-31)
        union FragU { uint4 u[2]; v16bf v; };
        FragU afrag[4], bfrag[2];
        #pragma unroll
        for (int i = 0; i < 4; ++i) {
            const int r = wm * 64 + i * 16 + lane16;
            afrag[i].u[0] = *(const uint4*)&Als[r * 40 + halfk];
            afrag[i].u[1] = *(const uint4*)&Als[r * 40 + 16 + halfk];
        }
        #pragma unroll
        for (int j = 0; j < 2; ++j) {
            const int r = wn * 32 + j * 16 + lane16;
            bfrag[j].u[0] = *(const uint4*)&Bls[r * 40 + halfk];
            bfrag[j].u[1] = *(const uint4*)&Bls[r * 40 + 16 + halfk];
        }

        #pragma unroll
        for (int i = 0; i < 4; ++i)
            #pragma unroll
            for (int j = 0; j < 2; ++j)
                acc[i][j] = __builtin_amdgcn_wmma_f32_16x16x32_bf16(
                    false, afrag[i].v, false, bfrag[j].v,
                    (short)0, acc[i][j], false, false);
        __syncthreads();
    }

    // ---- epilogue (C layout: VGPR r -> M = r + (lane>=16 ? 8 : 0), N = lane%16)
    const int r0 = (lane >> 4) << 3;
    #pragma unroll
    for (int i = 0; i < 4; ++i) {
        #pragma unroll
        for (int j = 0; j < 2; ++j) {
            const int gn = bn + wn * 32 + j * 16 + lane16;
            const float bs = bias ? bias[gn] : 0.f;
            #pragma unroll
            for (int r = 0; r < 8; ++r) {
                float v = acc[i][j][r] + bs;
                if (RELU) v = v > 0.f ? v : 0.f;
                const int gm = bm + wm * 64 + i * 16 + r0 + r;
                const size_t idx = TRANSPOSED ? (size_t)gn * ldc + gm
                                              : (size_t)gm * ldc + gn;
                if (OUT_BF16) ((unsigned short*)Cp)[idx] = f2bf(v);
                else          ((float*)Cp)[idx] = v;
            }
        }
    }
}

// ---------------------------------------------------------------------------
// Segment-mean pooling (batch sorted): block = graph, thread = feature
// ---------------------------------------------------------------------------
__device__ __forceinline__ int lower_bound_i(const int* __restrict__ b, int n, int v) {
    int lo = 0, hi = n;
    while (lo < hi) { int m = (lo + hi) >> 1; if (b[m] < v) lo = m + 1; else hi = m; }
    return lo;
}

__global__ void pool_kernel(const float* __restrict__ h2,
                            const int* __restrict__ batch,
                            float* __restrict__ pooled) {
    const int g = blockIdx.x;          // 64 graphs
    const int d = threadIdx.x;         // 256 features
    const int s = lower_bound_i(batch, N_NODES, g);
    const int e = lower_bound_i(batch, N_NODES, g + 1);
    float sum = 0.f;
    for (int i = s; i < e; ++i) sum += h2[(size_t)i * DOUT + d];
    int cnt = e - s; if (cnt < 1) cnt = 1;
    pooled[g * DOUT + d] = sum / (float)cnt;
}

// ---------------------------------------------------------------------------
// Tiny MLP head (f32 VALU; negligible work)
// ---------------------------------------------------------------------------
__global__ void head1_kernel(const float* __restrict__ pooled,
                             const float* __restrict__ Wc1,
                             const float* __restrict__ bc1,
                             float* __restrict__ z) {
    const int idx = blockIdx.x * 256 + threadIdx.x;   // 64 * 512
    const int g = idx >> 9, j = idx & 511;
    const float* p = pooled + g * DOUT;
    float s = bc1[j];
    for (int k = 0; k < DOUT; ++k) s += p[k] * Wc1[(size_t)k * DH + j];
    z[idx] = s > 0.f ? s : 0.f;
}

__global__ void head2_kernel(const float* __restrict__ z,
                             const float* __restrict__ Wc2,
                             const float* __restrict__ bc2,
                             float* __restrict__ out) {
    const int idx = blockIdx.x * 256 + threadIdx.x;
    if (idx >= NUM_GRAPHS * NUM_CLASSES) return;
    const int g = idx / NUM_CLASSES, c = idx % NUM_CLASSES;
    float s = bc2[c];
    for (int k = 0; k < DH; ++k) s += z[g * DH + k] * Wc2[(size_t)k * NUM_CLASSES + c];
    out[idx] = s;
}

// ---------------------------------------------------------------------------
extern "C" void kernel_launch(void* const* d_in, const int* in_sizes, int n_in,
                              void* d_out, int out_size, void* d_ws, size_t ws_size,
                              hipStream_t stream) {
    const float* x    = (const float*)d_in[0];
    const float* adj  = (const float*)d_in[1];
    const int*   batch= (const int*)  d_in[2];
    const float* W1   = (const float*)d_in[3];
    const float* b1   = (const float*)d_in[4];
    const float* W2   = (const float*)d_in[5];
    const float* b2   = (const float*)d_in[6];
    const float* Wc1  = (const float*)d_in[7];
    const float* bc1  = (const float*)d_in[8];
    const float* Wc2  = (const float*)d_in[9];
    const float* bc2  = (const float*)d_in[10];
    float* out = (float*)d_out;

    // workspace carve-out (256B aligned)
    char* w = (char*)d_ws;
    size_t off = 0;
    auto carve = [&](size_t bytes) -> void* {
        void* p = w + off;
        off = (off + bytes + 255) & ~(size_t)255;
        return p;
    };
    unsigned short* Abf = (unsigned short*)carve((size_t)N_NODES * N_NODES * 2); // 128 MiB
    unsigned short* Xb  = (unsigned short*)carve((size_t)N_NODES * DIN * 2);
    unsigned short* W1t = (unsigned short*)carve((size_t)DH * DIN * 2);
    unsigned short* W2t = (unsigned short*)carve((size_t)DOUT * DH * 2);
    unsigned short* T1t = (unsigned short*)carve((size_t)DH * N_NODES * 2);
    unsigned short* H1b = (unsigned short*)carve((size_t)N_NODES * DH * 2);
    unsigned short* T2t = (unsigned short*)carve((size_t)DOUT * N_NODES * 2);
    float* H2     = (float*)carve((size_t)N_NODES * DOUT * 4);
    float* dinv   = (float*)carve((size_t)N_NODES * 4);
    float* pooled = (float*)carve((size_t)NUM_GRAPHS * DOUT * 4);
    float* z      = (float*)carve((size_t)NUM_GRAPHS * DH * 4);

    // 1) degrees -> dinv
    rowsum_dinv_kernel<<<N_NODES, 256, 0, stream>>>(adj, dinv);
    // 2) normalized adjacency -> bf16 (one pass; 128 MiB result fits in L2)
    norm_cvt_kernel<<<(int)(((size_t)N_NODES * N_NODES / 4) / 256), 256, 0, stream>>>(adj, dinv, Abf);
    // 3) precision / layout prep
    cvt_bf16_kernel<<<(N_NODES * DIN / 4 + 255) / 256, 256, 0, stream>>>(x, Xb, N_NODES * DIN / 4);
    transpose_cvt_kernel<<<(DIN * DH + 255) / 256, 256, 0, stream>>>(W1, W1t, DIN, DH);   // W1t[n][k]
    transpose_cvt_kernel<<<(DH * DOUT + 255) / 256, 256, 0, stream>>>(W2, W2t, DH, DOUT); // W2t[n][k]

    // 4) T1 = X @ W1, stored transposed (so it is the next GEMM's Bt)
    gemm_wmma_kernel<true,  true,  false><<<dim3(N_NODES / 128, DH / 128),   256, 0, stream>>>(
        Xb, DIN, W1t, DIN, T1t, N_NODES, nullptr, DIN);
    // 5) H1 = relu(A @ T1 + b1), row-major bf16
    gemm_wmma_kernel<true,  false, true ><<<dim3(N_NODES / 128, DH / 128),   256, 0, stream>>>(
        Abf, N_NODES, T1t, N_NODES, H1b, DH, b1, N_NODES);
    // 6) T2 = H1 @ W2, stored transposed
    gemm_wmma_kernel<true,  true,  false><<<dim3(N_NODES / 128, DOUT / 128), 256, 0, stream>>>(
        H1b, DH, W2t, DH, T2t, N_NODES, nullptr, DH);
    // 7) H2 = A @ T2 + b2, f32
    gemm_wmma_kernel<false, false, false><<<dim3(N_NODES / 128, DOUT / 128), 256, 0, stream>>>(
        Abf, N_NODES, T2t, N_NODES, H2, DOUT, b2, N_NODES);

    // 8) mean pooling per graph
    pool_kernel<<<NUM_GRAPHS, DOUT, 0, stream>>>(H2, batch, pooled);
    // 9) classifier head
    head1_kernel<<<NUM_GRAPHS * DH / 256, 256, 0, stream>>>(pooled, Wc1, bc1, z);
    head2_kernel<<<(NUM_GRAPHS * NUM_CLASSES + 255) / 256, 256, 0, stream>>>(z, Wc2, bc2, out);
}
